// TopKMoeLayer_46840913330226
// MI455X (gfx1250) — matrix operand order repaired and checked by reading, not compiled, not run on hardware
//
#include <hip/hip_runtime.h>
#include <hip/hip_bf16.h>
#include <stdint.h>

// ---------------- problem constants (match reference) ----------------
constexpr int T_TOK  = 8192;   // 4 * 2048 tokens
constexpr int DIM    = 1024;   // model dim (square experts)
constexpr int NEXP   = 10;     // total experts (8 true + 2 null)
constexpr int NTRUE  = 8;

// ---------------- types ----------------
typedef __bf16 bf16;
typedef __attribute__((ext_vector_type(8)))  __bf16 bf16x8;
typedef __attribute__((ext_vector_type(16))) __bf16 bf16x16;
typedef __attribute__((ext_vector_type(8)))  float  floatx8;

static __device__ __forceinline__ floatx8 wmma_bf16(bf16x16 a, bf16x16 b, floatx8 c) {
  // D = A*B + C, 16x16x32 bf16 -> f32 accumulate (CDNA5 WMMA)
  return __builtin_amdgcn_wmma_f32_16x16x32_bf16(false, a, false, b, (short)0, c, false, false);
}

static __device__ __forceinline__ bf16x16 combine8(bf16x8 lo, bf16x8 hi) {
  return __builtin_shufflevector(lo, hi, 0, 1, 2, 3, 4, 5, 6, 7,
                                         8, 9, 10, 11, 12, 13, 14, 15);
}

// ============================================================================
// Kernel 1: split fp32 activations into bf16 hi / lo planes (one BW pass).
// x_lo = bf16(x - float(bf16(x))) so that x_hi + x_lo ~= x to fp32 accuracy.
// ============================================================================
__global__ __launch_bounds__(256) void moe_convert_x(
    const float* __restrict__ x, bf16* __restrict__ xhi, bf16* __restrict__ xlo) {
  size_t base = ((size_t)blockIdx.x * 256 + threadIdx.x) * 8;
  float4 a = *(const float4*)(x + base);
  float4 b = *(const float4*)(x + base + 4);
  float f[8] = {a.x, a.y, a.z, a.w, b.x, b.y, b.z, b.w};
  bf16x8 h, l;
#pragma unroll
  for (int j = 0; j < 8; ++j) {
    bf16 hv = (bf16)f[j];
    h[j] = hv;
    l[j] = (bf16)(f[j] - (float)hv);
  }
  *(bf16x8*)(xhi + base) = h;
  *(bf16x8*)(xlo + base) = l;
}

// ============================================================================
// Kernel 2: split + transpose expert weights: fp32 [e][k][n] ->
// bf16 hi/lo planes [e][n][k] (K-contiguous so B fragments / async copies are
// contiguous 16B chunks). LDS-tiled 64x64 transpose, coalesced on both sides.
// ============================================================================
constexpr int TT = 64;

__global__ __launch_bounds__(256) void moe_convert_w(
    const float* __restrict__ ew, bf16* __restrict__ whi, bf16* __restrict__ wlo) {
  __shared__ bf16 th[TT][TT + 4];
  __shared__ bf16 tl[TT][TT + 4];
  const int e = blockIdx.z, k0 = blockIdx.y * TT, n0 = blockIdx.x * TT;
  const float* src = ew + (size_t)e * DIM * DIM;
  const int tid = threadIdx.x;
#pragma unroll
  for (int i = 0; i < (TT * TT) / 256; ++i) {
    int idx = tid + i * 256;
    int k = idx >> 6, n = idx & (TT - 1);            // coalesced over n
    float v = src[(size_t)(k0 + k) * DIM + n0 + n];
    bf16 h = (bf16)v;
    th[n][k] = h;
    tl[n][k] = (bf16)(v - (float)h);
  }
  __syncthreads();
  bf16* dh = whi + (size_t)e * DIM * DIM;
  bf16* dl = wlo + (size_t)e * DIM * DIM;
#pragma unroll
  for (int i = 0; i < (TT * TT) / 256; ++i) {
    int idx = tid + i * 256;
    int n = idx >> 6, k = idx & (TT - 1);            // coalesced over k
    dh[(size_t)(n0 + n) * DIM + k0 + k] = th[n][k];
    dl[(size_t)(n0 + n) * DIM + k0 + k] = tl[n][k];
  }
}

// ============================================================================
// Kernel 3: gating. One wave32 per token: 10 gate dots (lane-parallel over D),
// softmax, top-2, null-expert mask + renormalize, atomic compaction into
// per-expert token lists.
// ============================================================================
__global__ __launch_bounds__(256) void moe_gate(
    const float* __restrict__ x, const float* __restrict__ gw,
    const float* __restrict__ gb, int* __restrict__ cnt,
    int* __restrict__ tidx, float* __restrict__ tw) {
  const int wave = threadIdx.x >> 5;
  const int lane = threadIdx.x & 31;
  const int t = blockIdx.x * 8 + wave;  // grid sized so t < T_TOK always

  const float* xr = x + (size_t)t * DIM;
  float xv[DIM / 32];
#pragma unroll
  for (int i = 0; i < DIM / 32; ++i) xv[i] = xr[lane + 32 * i];

  float logits[NEXP];
#pragma unroll
  for (int e = 0; e < NEXP; ++e) {
    const float* g = gw + (size_t)e * DIM;
    float s = 0.f;
#pragma unroll
    for (int i = 0; i < DIM / 32; ++i) s += xv[i] * g[lane + 32 * i];
#pragma unroll
    for (int off = 16; off > 0; off >>= 1) s += __shfl_xor(s, off, 32);
    logits[e] = s + gb[e];  // xor-reduce leaves full sum in every lane
  }

  if (lane == 0) {
    float m = logits[0];
#pragma unroll
    for (int e = 1; e < NEXP; ++e) m = fmaxf(m, logits[e]);
    float p[NEXP], sum = 0.f;
#pragma unroll
    for (int e = 0; e < NEXP; ++e) { p[e] = __expf(logits[e] - m); sum += p[e]; }
    // top-2 (first index wins ties, matching lax.top_k)
    int i1 = 0;
#pragma unroll
    for (int e = 1; e < NEXP; ++e) if (p[e] > p[i1]) i1 = e;
    int i2 = (i1 == 0) ? 1 : 0;
#pragma unroll
    for (int e = 0; e < NEXP; ++e) if (e != i1 && p[e] > p[i2]) i2 = e;
    float w1 = p[i1] / sum, w2 = p[i2] / sum;
    float denom = (i1 < NTRUE ? w1 : 0.f) + (i2 < NTRUE ? w2 : 0.f);
    if (denom > 0.f) {
      if (i1 < NTRUE) {
        int pos = atomicAdd(&cnt[i1], 1);
        tidx[i1 * T_TOK + pos] = t;
        tw[i1 * T_TOK + pos] = w1 / denom;
      }
      if (i2 < NTRUE) {
        int pos = atomicAdd(&cnt[i2], 1);
        tidx[i2 * T_TOK + pos] = t;
        tw[i2 * T_TOK + pos] = w2 / denom;
      }
    }
  }
}

// ============================================================================
// Kernel 4: per-expert gathered GEMM, bf16x3 split on WMMA.
// Block tile: M=128 gathered tokens x N=128 outputs, K-step 32.
// B tiles (pre-converted bf16, K-contiguous) are staged into double-buffered
// LDS with CDNA5 async copies (global_load_async_to_lds_b128 / ASYNCcnt),
// overlapping DMA of tile k+1 with WMMA compute of tile k.
// Epilogue: out[tok] += gate_w * (acc + bias) via global f32 atomics.
// ============================================================================
constexpr int BM = 128, BN = 128, BK = 32;
constexpr int KP = 40;  // LDS row pitch (elems): 80B rows, 16B-aligned, conflict-free

// Issue async copies of one 128x32 bf16 tile pair (hi+lo) into LDS buffers.
// Each of 256 threads moves 2 chunks of 16B per plane (512 chunks/plane).
static __device__ __forceinline__ void stage_async(
    uint32_t ldsh, uint32_t ldsl,
    const bf16* __restrict__ gh, const bf16* __restrict__ gl, int tid) {
#pragma unroll
  for (int j = 0; j < 2; ++j) {
    int id = tid + j * 256;
    int n = id >> 2;           // 0..127
    int c = id & 3;            // 16B chunk within the 64B row
    uint32_t loff = (uint32_t)(n * (KP * 2) + c * 16);
    const bf16* ph = gh + (size_t)n * DIM + c * 8;
    const bf16* pl = gl + (size_t)n * DIM + c * 8;
    asm volatile("global_load_async_to_lds_b128 %0, %1, off"
                 :: "v"(ldsh + loff), "v"(ph) : "memory");
    asm volatile("global_load_async_to_lds_b128 %0, %1, off"
                 :: "v"(ldsl + loff), "v"(pl) : "memory");
  }
}

__global__ __launch_bounds__(256) void moe_expert_gemm(
    const bf16* __restrict__ xhi, const bf16* __restrict__ xlo,
    const bf16* __restrict__ whi, const bf16* __restrict__ wlo,
    const float* __restrict__ eb,
    const int* __restrict__ cnt, const int* __restrict__ tidx,
    const float* __restrict__ tw, float* __restrict__ out) {
  const int e  = blockIdx.z;
  const int n0 = blockIdx.x * BN;
  const int m0 = blockIdx.y * BM;
  const int count = cnt[e];
  if (m0 >= count) return;  // uniform early-exit (most blocks)

  __shared__ bf16 Bh[2][BN * KP];
  __shared__ bf16 Bl[2][BN * KP];

  const int tid = threadIdx.x, lane = tid & 31, wave = tid >> 5;

  // Gathered A row for this lane (rows 16w..16w+15; lanes 16-31 mirror rows).
  const int mrow_base = m0 + wave * 16;
  const int arow = mrow_base + (lane & 15);
  const int tokenRow = (arow < count) ? tidx[e * T_TOK + arow] : 0;
  const bf16* ahp = xhi + (size_t)tokenRow * DIM;
  const bf16* alp = xlo + (size_t)tokenRow * DIM;
  const int aoff = (lane >> 4) * 8;  // 16-bit A layout: K chunks {0,16} / {8,24}

  floatx8 acc[8];
#pragma unroll
  for (int ci = 0; ci < 8; ++ci)
    acc[ci] = (floatx8){0.f, 0.f, 0.f, 0.f, 0.f, 0.f, 0.f, 0.f};

  const bf16* wh_base = whi + (size_t)e * DIM * DIM + (size_t)n0 * DIM;
  const bf16* wl_base = wlo + (size_t)e * DIM * DIM + (size_t)n0 * DIM;
  const uint32_t lh[2] = {(uint32_t)(uintptr_t)&Bh[0][0], (uint32_t)(uintptr_t)&Bh[1][0]};
  const uint32_t ll[2] = {(uint32_t)(uintptr_t)&Bl[0][0], (uint32_t)(uintptr_t)&Bl[1][0]};

  // Prologue: kick off DMA of the first B tile.
  stage_async(lh[0], ll[0], wh_base, wl_base, tid);

  for (int k0 = 0; k0 < DIM; k0 += BK) {
    const int b = (k0 / BK) & 1;
    // Drain this wave's outstanding async copies (current tile), then make
    // all waves' copies visible and all reads of the other buffer finished.
    asm volatile("s_wait_asynccnt 0x0" ::: "memory");
    __syncthreads();
    // Overlap: start DMA of the next tile into the other buffer.
    if (k0 + BK < DIM)
      stage_async(lh[1 - b], ll[1 - b], wh_base + (k0 + BK), wl_base + (k0 + BK), tid);

    // A fragments (hi/lo) straight from global (L2-resident bf16 planes).
    bf16x16 Ah = combine8(*(const bf16x8*)(ahp + k0 + aoff),
                          *(const bf16x8*)(ahp + k0 + aoff + 16));
    bf16x16 Al = combine8(*(const bf16x8*)(alp + k0 + aoff),
                          *(const bf16x8*)(alp + k0 + aoff + 16));

    const bf16* bh_buf = &Bh[b][0];
    const bf16* bl_buf = &Bl[b][0];
#pragma unroll
    for (int ci = 0; ci < 8; ++ci) {
      const int nl = ci * 16 + (lane & 15);
      const int bo = (lane >> 4) * 8;  // B layout: K chunks {0,16} / {8,24} (matches A)
      const bf16* ph = bh_buf + nl * KP + bo;
      const bf16* pl = bl_buf + nl * KP + bo;
      bf16x16 Bfh = combine8(*(const bf16x8*)ph, *(const bf16x8*)(ph + 16));
      bf16x16 Bfl = combine8(*(const bf16x8*)pl, *(const bf16x8*)(pl + 16));
      // fp32-grade product: Ah*Bh + Ah*Bl + Al*Bh
      acc[ci] = wmma_bf16(Ah, Bfh, acc[ci]);
      acc[ci] = wmma_bf16(Ah, Bfl, acc[ci]);
      acc[ci] = wmma_bf16(Al, Bfh, acc[ci]);
    }
  }

  // Epilogue: C layout -> lane holds col (lane&15); regs v are rows v or 8+v.
  const int col_lane = lane & 15;
  const int rsel = lane >> 4;
#pragma unroll
  for (int ci = 0; ci < 8; ++ci) {
    const int col = n0 + ci * 16 + col_lane;
    const float bias = eb[(size_t)e * DIM + col];
#pragma unroll
    for (int v = 0; v < 8; ++v) {
      const int m = mrow_base + rsel * 8 + v;
      if (m < count) {
        const float w = tw[e * T_TOK + m];
        const int trow = tidx[e * T_TOK + m];
        atomicAdd(&out[(size_t)trow * DIM + col], w * (acc[ci][v] + bias));
      }
    }
  }
}

// ============================================================================
// Launch
// ============================================================================
extern "C" void kernel_launch(void* const* d_in, const int* in_sizes, int n_in,
                              void* d_out, int out_size, void* d_ws, size_t ws_size,
                              hipStream_t stream) {
  const float* x  = (const float*)d_in[0];  // [T, D]
  const float* gw = (const float*)d_in[1];  // [10, D]
  const float* gb = (const float*)d_in[2];  // [10]
  const float* ew = (const float*)d_in[3];  // [8, D, D]
  const float* eb = (const float*)d_in[4];  // [8, D]
  float* out = (float*)d_out;               // [T, D]

  // Workspace layout (~65 MB):
  //   counters | token lists | x bf16 hi/lo (32 MB) | w^T bf16 hi/lo (32 MB)
  char* ws = (char*)d_ws;
  int*   cnt  = (int*)ws;                                   // 8 ints
  int*   tidx = (int*)(ws + 1024);                          // 8*T ints (256 KB)
  float* tw   = (float*)(ws + 1024 + NTRUE * T_TOK * 4);    // 8*T floats (256 KB)
  bf16*  xhi  = (bf16*)(ws + (1ull << 20));                 // 16 MB
  bf16*  xlo  = xhi + (size_t)T_TOK * DIM;                  // 16 MB
  bf16*  whi  = (bf16*)(ws + (33ull << 20));                // 16 MB
  bf16*  wlo  = whi + (size_t)NTRUE * DIM * DIM;            // 16 MB

  hipMemsetAsync(cnt, 0, NTRUE * sizeof(int), stream);
  hipMemsetAsync(out, 0, (size_t)out_size * sizeof(float), stream);

  // 1) fp32 -> bf16 hi/lo split of activations
  moe_convert_x<<<(T_TOK * DIM) / (256 * 8), 256, 0, stream>>>(x, xhi, xlo);

  // 2) fp32 -> bf16 hi/lo split + transpose of expert weights ([e][n][k])
  dim3 wgrid(DIM / TT, DIM / TT, NTRUE);
  moe_convert_w<<<wgrid, 256, 0, stream>>>(ew, whi, wlo);

  // 3) gating + per-expert token compaction (1 wave per token)
  moe_gate<<<T_TOK / 8, 256, 0, stream>>>(x, gw, gb, cnt, tidx, tw);

  // 4) gathered expert GEMMs: grid = (N tiles, worst-case M tiles, experts)
  dim3 grid(DIM / BN, T_TOK / BM, NTRUE);
  moe_expert_gemm<<<grid, 256, 0, stream>>>(xhi, xlo, whi, wlo, eb,
                                            cnt, tidx, tw, out);
}
